// ToMeBlock_52278341927303
// MI455X (gfx1250) — matrix operand order repaired and checked by reading, not compiled, not run on hardware
//
#include <hip/hip_runtime.h>
#include <hip/hip_bf16.h>

// ---------------- problem constants (from reference) ----------------
#define BATCH 16
#define HH    64
#define WW    64
#define NTOK  4096          // H*W
#define CH    320
#define NSRC  3072          // |A_IDX|
#define NDST  1024          // |B_IDX|
#define RMERGE 2048         // r = int(N * 0.5)
#define SORTN 4096          // bitonic pad

typedef __attribute__((ext_vector_type(16))) _Float16 v16h;
typedef __attribute__((ext_vector_type(8)))  _Float16 v8h;
typedef __attribute__((ext_vector_type(8)))  float    v8f;

// dst token j -> pixel index (even row, even col, row-major over 32x32 blocks)
__device__ __forceinline__ int b_index(int j) {
    int by = j >> 5, bx = j & 31;
    return (by * 2) * WW + bx * 2;
}
// src token i (0..3071) -> pixel index (row-major pixels excluding even/even)
__device__ __forceinline__ int a_index(int i) {
    int g = i / 96, r = i % 96;   // 2-row group: 32 (odd-x of even row) + 64 (odd row)
    int y, x;
    if (r < 32) { y = 2 * g;     x = 2 * r + 1; }
    else        { y = 2 * g + 1; x = r - 32;    }
    return y * WW + x;
}

// ---------------- K1: normalize + pack to f16 (wave per token row) ----------------
__global__ __launch_bounds__(256)
void tome_pack_kernel(const float* __restrict__ x,
                      _Float16* __restrict__ pA, _Float16* __restrict__ pB) {
    int rowLin = blockIdx.x * 8 + (threadIdx.x >> 5);
    int lane   = threadIdx.x & 31;
    int b   = rowLin >> 12;
    int tok = rowLin & (NTOK - 1);
    int n; _Float16* dst;
    if (tok < NSRC) { n = a_index(tok);        dst = pA + ((size_t)b * NSRC + tok) * CH; }
    else            { int d = tok - NSRC; n = b_index(d);
                      dst = pB + ((size_t)b * NDST + d) * CH; }
    const float* src = x + ((size_t)b * NTOK + n) * CH;
    float v[10]; float ss = 0.f;
#pragma unroll
    for (int i = 0; i < 10; ++i) { v[i] = src[lane + 32 * i]; ss += v[i] * v[i]; }
#pragma unroll
    for (int m = 16; m >= 1; m >>= 1) ss += __shfl_xor(ss, m, 32);
    float inv = rsqrtf(ss);
#pragma unroll
    for (int i = 0; i < 10; ++i) dst[lane + 32 * i] = (_Float16)(v[i] * inv);
}

// ---------------- K2: fused WMMA GEMM + row max/argmax ----------------
// block = 128 threads (4 waves). block -> (batch, 32-row src tile pair).
// Each wave holds TWO register-resident A tiles (rows t*32+0..15 and +16..31)
// and issues 2 WMMAs per B-fragment load -> halves L2 B traffic.
// wave w handles dst tiles {w, w+4, ...}: 16 tiles x 20 WMMA (K=320).
__global__ __launch_bounds__(128)
void tome_gemm_argmax_kernel(const _Float16* __restrict__ pA,
                             const _Float16* __restrict__ pB,
                             float* __restrict__ node_max,
                             int*   __restrict__ node_idx) {
    const int b    = blockIdx.x / (NSRC / 32);
    const int t    = blockIdx.x % (NSRC / 32);
    const int wave = threadIdx.x >> 5;
    const int lane = threadIdx.x & 31;
    const int hs   = lane >> 4;    // K-half select
    const int l16  = lane & 15;

    __shared__ float sMax[4][32];
    __shared__ int   sIdx[4][32];

    // Preload all 10 A fragments for BOTH 16-row tiles.
    // A 16x32 f16 layout: lane half 0 -> K {0..7,16..23}+32kk ; half 1 -> +8.
    const _Float16* aRow0 = pA + ((size_t)b * NSRC + t * 32 + l16) * CH;
    const _Float16* aRow1 = aRow0 + (size_t)16 * CH;
    v16h aFragLo[10], aFragHi[10];
#pragma unroll
    for (int kk = 0; kk < 10; ++kk) {
        int k0 = kk * 32 + hs * 8;
        v8h c0 = *(const v8h*)(aRow0 + k0);
        v8h c1 = *(const v8h*)(aRow0 + k0 + 16);
        v8h d0 = *(const v8h*)(aRow1 + k0);
        v8h d1 = *(const v8h*)(aRow1 + k0 + 16);
#pragma unroll
        for (int i = 0; i < 8; ++i) {
            aFragLo[kk][i] = c0[i]; aFragLo[kk][8 + i] = c1[i];
            aFragHi[kk][i] = d0[i]; aFragHi[kk][8 + i] = d1[i];
        }
    }

    float bestLo[8], bestHi[8]; int bidxLo[8], bidxHi[8];
#pragma unroll
    for (int j = 0; j < 8; ++j) {
        bestLo[j] = -3.0e38f; bidxLo[j] = 0x7fffffff;
        bestHi[j] = -3.0e38f; bidxHi[j] = 0x7fffffff;
    }

    for (int dt = wave; dt < NDST / 16; dt += 4) {
        // B 32x16 f16 layout: lane = col (l16); half 0 -> K {0..7,16..23}+32kk ; half 1 -> +8.
        const _Float16* bRow = pB + ((size_t)b * NDST + dt * 16 + l16) * CH;
        v8f accLo = {}, accHi = {};
#pragma unroll
        for (int kk = 0; kk < 10; ++kk) {
            int k0 = kk * 32 + hs * 8;
            v8h c0 = *(const v8h*)(bRow + k0);
            v8h c1 = *(const v8h*)(bRow + k0 + 16);
            v16h bFrag;
#pragma unroll
            for (int i = 0; i < 8; ++i) { bFrag[i] = c0[i]; bFrag[8 + i] = c1[i]; }
            accLo = __builtin_amdgcn_wmma_f32_16x16x32_f16(
                false, aFragLo[kk], false, bFrag, (short)0, accLo, false, false);
            accHi = __builtin_amdgcn_wmma_f32_16x16x32_f16(
                false, aFragHi[kk], false, bFrag, (short)0, accHi, false, false);
        }
        int col = dt * 16 + l16;            // C/D: lane -> N=l16, elem j -> M=j+8*hs
#pragma unroll
        for (int j = 0; j < 8; ++j) {
            float v0 = accLo[j];
            if (v0 > bestLo[j] || (v0 == bestLo[j] && col < bidxLo[j])) { bestLo[j] = v0; bidxLo[j] = col; }
            float v1 = accHi[j];
            if (v1 > bestHi[j] || (v1 == bestHi[j] && col < bidxHi[j])) { bestHi[j] = v1; bidxHi[j] = col; }
        }
    }
    // reduce across the 16 lanes of each half (same M, different N)
#pragma unroll
    for (int m = 1; m < 16; m <<= 1) {
#pragma unroll
        for (int j = 0; j < 8; ++j) {
            float ov = __shfl_xor(bestLo[j], m, 32);
            int   oi = __shfl_xor(bidxLo[j], m, 32);
            if (ov > bestLo[j] || (ov == bestLo[j] && oi < bidxLo[j])) { bestLo[j] = ov; bidxLo[j] = oi; }
            float ow = __shfl_xor(bestHi[j], m, 32);
            int   oj = __shfl_xor(bidxHi[j], m, 32);
            if (ow > bestHi[j] || (ow == bestHi[j] && oj < bidxHi[j])) { bestHi[j] = ow; bidxHi[j] = oj; }
        }
    }
    if (l16 == 0) {
#pragma unroll
        for (int j = 0; j < 8; ++j) {
            int row = j + hs * 8;
            sMax[wave][row]      = bestLo[j]; sIdx[wave][row]      = bidxLo[j];
            sMax[wave][16 + row] = bestHi[j]; sIdx[wave][16 + row] = bidxHi[j];
        }
    }
    __syncthreads();
    if (threadIdx.x < 32) {
        int row = threadIdx.x;
        float v = sMax[0][row]; int ix = sIdx[0][row];
#pragma unroll
        for (int w2 = 1; w2 < 4; ++w2) {
            float ov = sMax[w2][row]; int oi = sIdx[w2][row];
            if (ov > v || (ov == v && oi < ix)) { v = ov; ix = oi; }
        }
        int g = t * 32 + row;
        node_max[(size_t)b * NSRC + g] = v;
        node_idx[(size_t)b * NSRC + g] = ix;
    }
}

// ---------------- K3: per-batch bitonic argsort (desc score, asc idx) ----------------
__global__ __launch_bounds__(1024)
void tome_sort_kernel(const float* __restrict__ node_max, int* __restrict__ sorted_idx) {
    __shared__ unsigned long long key[SORTN];
    int b = blockIdx.x;
    for (int i = threadIdx.x; i < SORTN; i += 1024) {
        unsigned long long k;
        if (i < NSRC) {
            unsigned u = __float_as_uint(node_max[(size_t)b * NSRC + i]);
            unsigned s = (u & 0x80000000u) ? ~u : (u | 0x80000000u); // ascending-orderable
            k = ((unsigned long long)(~s) << 32) | (unsigned)i;      // ~s: descending score
        } else {
            k = 0xFFFFFFFF00000000ull | (unsigned)i;                 // pad sorts last
        }
        key[i] = k;
    }
    __syncthreads();
    for (unsigned kk = 2; kk <= SORTN; kk <<= 1) {
        for (unsigned j = kk >> 1; j > 0; j >>= 1) {
            for (unsigned t = threadIdx.x; t < SORTN / 2; t += 1024) {
                unsigned i = ((t & ~(j - 1)) << 1) | (t & (j - 1));
                unsigned l = i + j;
                bool up = ((i & kk) == 0);
                unsigned long long a = key[i], c = key[l];
                if ((a > c) == up) { key[i] = c; key[l] = a; }
            }
            __syncthreads();
        }
    }
    for (int i = threadIdx.x; i < NSRC; i += 1024)
        sorted_idx[(size_t)b * NSRC + i] = (int)(key[i] & 0xFFFFFFFFu);
}

// ---------------- K4a: sums = x[dst rows], cnt = 1 ----------------
__global__ __launch_bounds__(256)
void tome_init_sums_kernel(const float* __restrict__ x,
                           float* __restrict__ sums, float* __restrict__ cnt) {
    int idx = blockIdx.x * 256 + threadIdx.x;
    if (idx >= BATCH * NDST * CH) return;
    int c = idx % CH; int bd = idx / CH;
    int d = bd % NDST; int b = bd / NDST;
    sums[idx] = x[((size_t)b * NTOK + b_index(d)) * CH + c];
    if (c == 0) cnt[bd] = 1.0f;
}

// ---------------- K4b: atomic scatter-add of top-r src rows ----------------
__global__ __launch_bounds__(320)
void tome_scatter_kernel(const float* __restrict__ x,
                         const int* __restrict__ sorted_idx,
                         const int* __restrict__ node_idx,
                         float* __restrict__ sums, float* __restrict__ cnt) {
    int e = blockIdx.x % RMERGE;
    int b = blockIdx.x / RMERGE;
    int s = sorted_idx[(size_t)b * NSRC + e];
    int d = node_idx[(size_t)b * NSRC + s];
    int n = a_index(s);
    int c = threadIdx.x;
    atomicAdd(&sums[((size_t)b * NDST + d) * CH + c],
              x[((size_t)b * NTOK + n) * CH + c]);
    if (c == 0) atomicAdd(&cnt[(size_t)b * NDST + d], 1.0f);
}

// ---------------- K5a: dst_m = sums/cnt, write to dst positions of out ----------------
__global__ __launch_bounds__(256)
void tome_finalize_dst_kernel(float* __restrict__ sums, const float* __restrict__ cnt,
                              float* __restrict__ out) {
    int idx = blockIdx.x * 256 + threadIdx.x;
    if (idx >= BATCH * NDST * CH) return;
    int c = idx % CH; int bd = idx / CH;
    int d = bd % NDST; int b = bd / NDST;
    float v = sums[idx] / cnt[bd];
    sums[idx] = v;                                   // sums now holds dst_m
    out[((size_t)b * NTOK + b_index(d)) * CH + c] = v;
}

// ---------------- K5b: fill src positions (merged -> dst_m gather, unmerged -> x) ----------------
__global__ __launch_bounds__(320)
void tome_unmerge_kernel(const float* __restrict__ x,
                         const int* __restrict__ sorted_idx,
                         const int* __restrict__ node_idx,
                         const float* __restrict__ dstm, float* __restrict__ out) {
    int e = blockIdx.x % NSRC;
    int b = blockIdx.x / NSRC;
    int s = sorted_idx[(size_t)b * NSRC + e];
    int n = a_index(s);
    int c = threadIdx.x;
    float v;
    if (e < RMERGE) {
        int d = node_idx[(size_t)b * NSRC + s];
        v = dstm[((size_t)b * NDST + d) * CH + c];
    } else {
        v = x[((size_t)b * NTOK + n) * CH + c];
    }
    out[((size_t)b * NTOK + n) * CH + c] = v;
}

// ---------------- host launch ----------------
extern "C" void kernel_launch(void* const* d_in, const int* in_sizes, int n_in,
                              void* d_out, int out_size, void* d_ws, size_t ws_size,
                              hipStream_t stream) {
    (void)in_sizes; (void)n_in; (void)out_size; (void)ws_size;
    const float* x = (const float*)d_in[0];
    float* out = (float*)d_out;

    char* ws = (char*)d_ws;
    size_t off = 0;
    auto carve = [&](size_t bytes) -> char* {
        char* p = ws + off;
        off += (bytes + 255) & ~(size_t)255;
        return p;
    };
    _Float16* pA       = (_Float16*)carve((size_t)BATCH * NSRC * CH * sizeof(_Float16)); // 31.5 MB
    _Float16* pB       = (_Float16*)carve((size_t)BATCH * NDST * CH * sizeof(_Float16)); // 10.5 MB
    float*    node_max = (float*)   carve((size_t)BATCH * NSRC * sizeof(float));
    int*      node_idx = (int*)     carve((size_t)BATCH * NSRC * sizeof(int));
    int*      sortedix = (int*)     carve((size_t)BATCH * NSRC * sizeof(int));
    float*    sums     = (float*)   carve((size_t)BATCH * NDST * CH * sizeof(float));    // 21 MB
    float*    cnt      = (float*)   carve((size_t)BATCH * NDST * sizeof(float));

    // K1: normalize + pack (8 waves/block, one wave per token row)
    tome_pack_kernel<<<BATCH * NTOK / 8, 256, 0, stream>>>(x, pA, pB);

    // K2: fused WMMA GEMM + argmax. 16 batches x 96 row-tile pairs (32 rows each).
    tome_gemm_argmax_kernel<<<BATCH * (NSRC / 32), 128, 0, stream>>>(pA, pB, node_max, node_idx);

    // K3: per-batch bitonic sort
    tome_sort_kernel<<<BATCH, 1024, 0, stream>>>(node_max, sortedix);

    // K4: scatter-mean
    int tot = BATCH * NDST * CH;
    tome_init_sums_kernel<<<(tot + 255) / 256, 256, 0, stream>>>(x, sums, cnt);
    tome_scatter_kernel<<<BATCH * RMERGE, 320, 0, stream>>>(x, sortedix, node_idx, sums, cnt);

    // K5: finalize dst + unmerge src
    tome_finalize_dst_kernel<<<(tot + 255) / 256, 256, 0, stream>>>(sums, cnt, out);
    tome_unmerge_kernel<<<BATCH * NSRC, 320, 0, stream>>>(x, sortedix, node_idx, sums, out);
}
